// InstanceCommClusterPoints_29815662969430
// MI455X (gfx1250) — matrix operand order repaired and checked by reading, not compiled, not run on hardware
//
#include <hip/hip_runtime.h>
#include <hip/hip_bf16.h>
#include <stdint.h>

// ---------------------------------------------------------------------------
// Problem constants (from the reference)
// ---------------------------------------------------------------------------
#define NPTS      500000
#define NCLUST    40000
#define C2        (NCLUST * 2)       // (cluster, fg/bg) bins
#define D_FEAT    64
#define D_LOG     20
#define D_PTS     4
#define D_PROJ    3
#define D_RAW     91                 // 64+20+4+3
#define D_IN      182                // 2*D_RAW
#define K_PAD     192                // D_IN padded to 6 * 32 for bf16 WMMA
#define D_OUT     128
#define EPS       1e-5f
#define SLOPE     0.1f
#define TILE_PTS  128                // points per block (8 waves x 16 rows)

typedef __attribute__((ext_vector_type(16))) __bf16 v16bf;
typedef __attribute__((ext_vector_type(8)))  float  v8f;

union FragBF {            // 8 VGPRs of packed bf16 pairs
    v16bf v;
    uint4 q[2];
};

// ---------------------------------------------------------------------------
// Kernel 1: scatter-add per-(cluster, fg/bg) sums + counts.
// ws sums layout: [C2][92]  (91 feature sums in order feat|log|pts|proj, then count)
// ---------------------------------------------------------------------------
__global__ void scatter_sums_kernel(const float* __restrict__ feat,
                                    const float* __restrict__ logit,
                                    const float* __restrict__ pts,
                                    const float* __restrict__ proj,
                                    const int*   __restrict__ idx,
                                    const unsigned char* __restrict__ fg,
                                    float* __restrict__ sums, int n)
{
    long long total = (long long)n * 92;
    for (long long i = (long long)blockIdx.x * blockDim.x + threadIdx.x;
         i < total; i += (long long)gridDim.x * blockDim.x) {
        int p = (int)(i / 92);
        int j = (int)(i % 92);
        float v;
        if (j < 64)       v = feat [(long long)p * D_FEAT + j];
        else if (j < 84)  v = logit[(long long)p * D_LOG  + (j - 64)];
        else if (j < 88)  v = pts  [(long long)p * D_PTS  + (j - 84)];
        else if (j < 91)  v = proj [(long long)p * D_PROJ + (j - 88)];
        else              v = 1.0f;                       // count slot
        int e = idx[p] * 2 + (fg[p] ? 0 : 1);
        atomicAdd(&sums[(long long)e * 92 + j], v);
    }
}

// ---------------------------------------------------------------------------
// Kernel 2: sums -> means  (means layout: [C2][91])
// ---------------------------------------------------------------------------
__global__ void means_kernel(const float* __restrict__ sums,
                             float* __restrict__ means)
{
    long long total = (long long)C2 * D_RAW;
    for (long long i = (long long)blockIdx.x * blockDim.x + threadIdx.x;
         i < total; i += (long long)gridDim.x * blockDim.x) {
        int e = (int)(i / D_RAW);
        int j = (int)(i % D_RAW);
        float cnt = sums[(long long)e * 92 + 91];
        means[i] = (cnt > 0.0f) ? sums[(long long)e * 92 + j] / cnt : 0.0f;
    }
}

// ---------------------------------------------------------------------------
// Kernel 3: fused gather + GEMM (bf16 WMMA) + bias + BN-stat accumulation.
//   Each block: 128 points. W staged [n][k] bf16 (k padded to 192),
//   x tile staged [p][k] bf16. Wave w computes rows w*16..w*16+15 for all
//   128 output channels: 8 f32 accumulators, 6 k-steps -> 48 WMMAs / wave.
// ---------------------------------------------------------------------------
__global__ void
__launch_bounds__(256)
fused_gemm_kernel(const float* __restrict__ feat,
                  const float* __restrict__ logit,
                  const float* __restrict__ pts,
                  const float* __restrict__ proj,
                  const int*   __restrict__ idx,
                  const unsigned char* __restrict__ fg,
                  const float* __restrict__ W,
                  const float* __restrict__ bias,
                  const float* __restrict__ means,
                  float* __restrict__ h_out,
                  float* __restrict__ gStats,   // [0..127]=sum, [128..255]=sumsq
                  int n)
{
    __shared__ __align__(32) __bf16 sW[D_OUT * K_PAD];     // 48 KB, [n][k]
    __shared__ __align__(32) __bf16 sX[TILE_PTS * K_PAD];  // 48 KB, [p][k]
    __shared__ float sSum[D_OUT];
    __shared__ float sSq [D_OUT];
    __shared__ int   sMB [TILE_PTS];                       // mean base per point

    const int tid        = threadIdx.x;
    const int blockStart = blockIdx.x * TILE_PTS;

    // ---- stage W as bf16, transposition-free ([n][k], zero-padded k>=182)
    for (int i = tid; i < D_OUT * K_PAD; i += 256) {
        int nrow = i / K_PAD, k = i % K_PAD;
        float w = (k < D_IN) ? W[nrow * D_IN + k] : 0.0f;
        sW[i] = (__bf16)w;
    }
    if (tid < D_OUT) { sSum[tid] = 0.0f; sSq[tid] = 0.0f; }
    if (tid < TILE_PTS) {
        int p = blockStart + tid;
        int e = (p < n) ? (idx[p] * 2 + (fg[p] ? 0 : 1)) : 0;
        sMB[tid] = e * D_RAW;
    }
    __syncthreads();

    // ---- build x tile: [raw(91) | cluster-mean(91) | pad(10)]
    for (int i = tid; i < TILE_PTS * K_PAD; i += 256) {
        int lp = i / K_PAD, k = i % K_PAD;
        int p  = blockStart + lp;
        float x = 0.0f;
        if (p < n && k < D_IN) {
            if (k < D_RAW) {
                if (k < 64)      x = feat [(long long)p * D_FEAT + k];
                else if (k < 84) x = logit[(long long)p * D_LOG  + (k - 64)];
                else if (k < 88) x = pts  [(long long)p * D_PTS  + (k - 84)];
                else             x = proj [(long long)p * D_PROJ + (k - 88)];
            } else {
                x = means[(long long)sMB[lp] + (k - D_RAW)];
            }
        }
        sX[i] = (__bf16)x;
    }
    __syncthreads();

    const int wave = tid >> 5;
    const int lane = tid & 31;
    const int half = lane >> 4;
    const int ln   = lane & 15;     // A row / B-C column within tile

    // accumulators preloaded with bias (C value at (M,N) depends on N only)
    v8f acc[8];
#pragma unroll
    for (int nt = 0; nt < 8; ++nt) {
        float bb = bias[nt * 16 + ln];
        acc[nt] = (v8f){bb, bb, bb, bb, bb, bb, bb, bb};
    }

    const int arow = wave * 16 + ln;
#pragma unroll
    for (int kc = 0; kc < 6; ++kc) {
        // A fragment (16-bit A 16x32): VGPR v pair K =
        //   half*16 + v*2  -> one contiguous 32B LDS read (matches B).
        FragBF a;
        const uint4* ap =
            (const uint4*)&sX[arow * K_PAD + kc * 32 + half * 16];
        a.q[0] = ap[0];          // K = kc*32 + half*16 .. +7
        a.q[1] = ap[1];          // K = kc*32 + half*16 + 8 .. +15
#pragma unroll
        for (int nt = 0; nt < 8; ++nt) {
            // B fragment (32x16): lanes 0-15 K=0..15, lanes 16-31 K=16..31,
            // VGPR v pair K = half*16 + v*2 -> one contiguous 32B read.
            FragBF b;
            const uint4* bp =
                (const uint4*)&sW[(nt * 16 + ln) * K_PAD + kc * 32 + half * 16];
            b.q[0] = bp[0];
            b.q[1] = bp[1];
            acc[nt] = __builtin_amdgcn_wmma_f32_16x16x32_bf16(
                false, a.v, false, b.v, (short)0, acc[nt], false, false);
        }
    }

    // ---- epilogue: store h, accumulate per-channel sum / sumsq
#pragma unroll
    for (int nt = 0; nt < 8; ++nt) {
        int ch = nt * 16 + ln;
        float s = 0.0f, s2 = 0.0f;
#pragma unroll
        for (int v = 0; v < 8; ++v) {
            int row = v + 8 * half;                 // C/D: M = v + 8*half
            int p   = blockStart + wave * 16 + row;
            if (p < n) {
                float val = acc[nt][v];
                h_out[(size_t)p * D_OUT + ch] = val;
                s  += val;
                s2 += val * val;
            }
        }
        atomicAdd(&sSum[ch], s);     // ds_add_f32
        atomicAdd(&sSq [ch], s2);
    }
    __syncthreads();
    if (tid < D_OUT) {
        atomicAdd(&gStats[tid],         sSum[tid]);
        atomicAdd(&gStats[D_OUT + tid], sSq [tid]);
    }
}

// ---------------------------------------------------------------------------
// Kernel 4: per-channel BN coefficients  y = h*a + c
// ---------------------------------------------------------------------------
__global__ void coef_kernel(const float* __restrict__ gStats,
                            const float* __restrict__ gamma,
                            const float* __restrict__ beta,
                            float* __restrict__ coefs, float invN)
{
    int c = threadIdx.x;
    if (c < D_OUT) {
        float mean = gStats[c] * invN;
        float var  = gStats[D_OUT + c] * invN - mean * mean;
        float a    = gamma[c] * rsqrtf(var + EPS);
        coefs[c]         = a;
        coefs[D_OUT + c] = beta[c] - mean * a;
    }
}

// ---------------------------------------------------------------------------
// Kernel 5: normalize + LeakyReLU, in place on d_out
// ---------------------------------------------------------------------------
__global__ void bn_leaky_kernel(float* __restrict__ h,
                                const float* __restrict__ coefs,
                                long long total)
{
    __shared__ float sc[2 * D_OUT];
    if (threadIdx.x < 2 * D_OUT) sc[threadIdx.x] = coefs[threadIdx.x];
    __syncthreads();
    for (long long i = (long long)blockIdx.x * blockDim.x + threadIdx.x;
         i < total; i += (long long)gridDim.x * blockDim.x) {
        int c = (int)(i & (D_OUT - 1));
        float y = h[i] * sc[c] + sc[D_OUT + c];
        h[i] = (y >= 0.0f) ? y : SLOPE * y;
    }
}

// ---------------------------------------------------------------------------
// Host launcher (graph-capture safe: only async stream ops)
// ---------------------------------------------------------------------------
extern "C" void kernel_launch(void* const* d_in, const int* in_sizes, int n_in,
                              void* d_out, int out_size, void* d_ws, size_t ws_size,
                              hipStream_t stream)
{
    const float* pts   = (const float*)d_in[0];
    const float* proj  = (const float*)d_in[1];
    const float* feat  = (const float*)d_in[2];
    const float* logit = (const float*)d_in[3];
    const int*   idx   = (const int*)d_in[4];
    const unsigned char* fg = (const unsigned char*)d_in[5];  // jax bool = 1 byte
    const float* W     = (const float*)d_in[6];
    const float* bias  = (const float*)d_in[7];
    const float* gamma = (const float*)d_in[8];
    const float* beta  = (const float*)d_in[9];
    float* out = (float*)d_out;

    const int n = in_sizes[0] / D_PTS;   // 500000

    // workspace layout (floats)
    float* sums   = (float*)d_ws;                        // C2*92
    float* means  = sums  + (size_t)C2 * 92;             // C2*91
    float* gStats = means + (size_t)C2 * D_RAW;          // 256
    float* coefs  = gStats + 2 * D_OUT;                  // 256

    hipMemsetAsync(sums,   0, (size_t)C2 * 92 * sizeof(float), stream);
    hipMemsetAsync(gStats, 0, 2 * D_OUT * sizeof(float), stream);

    // 1) scatter sums
    {
        long long total = (long long)n * 92;
        int blocks = (int)((total + 255) / 256);
        if (blocks > 16384) blocks = 16384;
        scatter_sums_kernel<<<blocks, 256, 0, stream>>>(
            feat, logit, pts, proj, idx, fg, sums, n);
    }
    // 2) means
    {
        long long total = (long long)C2 * D_RAW;
        int blocks = (int)((total + 255) / 256);
        if (blocks > 8192) blocks = 8192;
        means_kernel<<<blocks, 256, 0, stream>>>(sums, means);
    }
    // 3) fused gather + WMMA GEMM + stats
    {
        int blocks = (n + TILE_PTS - 1) / TILE_PTS;      // 3907
        fused_gemm_kernel<<<blocks, 256, 0, stream>>>(
            feat, logit, pts, proj, idx, fg, W, bias, means, out, gStats, n);
    }
    // 4) BN coefficients
    coef_kernel<<<1, 128, 0, stream>>>(gStats, gamma, beta, coefs,
                                       1.0f / (float)n);
    // 5) normalize + LeakyReLU
    {
        long long total = (long long)n * D_OUT;
        bn_leaky_kernel<<<4096, 256, 0, stream>>>(out, coefs, total);
    }
}
